// MaskedDenseLayerMultiMasks_34797825032869
// MI455X (gfx1250) — matrix operand order repaired and checked by reading, not compiled, not run on hardware
//
#include <hip/hip_runtime.h>

typedef __attribute__((ext_vector_type(4)))  float        v4f;
typedef __attribute__((ext_vector_type(8)))  float        v8f;
typedef __attribute__((ext_vector_type(4)))  int          v4i;
typedef __attribute__((ext_vector_type(4)))  unsigned int v4u;
typedef __attribute__((ext_vector_type(16))) __bf16       v16bf;
typedef __attribute__((ext_vector_type(4)))  __bf16       v4bf;

#define BATCH   512
#define NMASK   16
#define IN_DIM  1024
#define OUT_DIM 1024

#define BM 128
#define BN 128
#define BK 32
#define AFS 36          // fp32 per A row in LDS (144B; 16B-aligned, conflict-free frag reads)
#define BS  40          // bf16 per B col in LDS (80B; 16B-aligned, conflict-free frag reads)
#define NPANEL (IN_DIM / BK)

#if __has_builtin(__builtin_amdgcn_global_load_async_to_lds_b128)
#define HAVE_ASYNC 1
#else
#define HAVE_ASYNC 0
#endif

#define AS1 __attribute__((address_space(1)))
#define AS3 __attribute__((address_space(3)))

typedef AS1 v4i* gv4p;   // global-AS 16B vector pointer
typedef AS3 v4i* lv4p;   // LDS-AS 16B vector pointer

union Frag  { v16bf bf; v4u u[2]; };
union Pack8 { v4u u; __bf16 b[8]; };

// Split 8 consecutive fp32 (in LDS) into bf16 hi / bf16 lo 16B chunks.
__device__ __forceinline__ void split8(const float* s, v4u& hi, v4u& lo) {
  v4f f0 = *(const v4f*)s;
  v4f f1 = *(const v4f*)(s + 4);
  Pack8 H, L;
#pragma unroll
  for (int t = 0; t < 4; ++t) {
    __bf16 hb = (__bf16)f0[t];
    H.b[t] = hb;
    L.b[t] = (__bf16)(f0[t] - (float)hb);
  }
#pragma unroll
  for (int t = 0; t < 4; ++t) {
    __bf16 hb = (__bf16)f1[t];
    H.b[4 + t] = hb;
    L.b[4 + t] = (__bf16)(f1[t] - (float)hb);
  }
  hi = H.u;
  lo = L.u;
}

// Stage one 128x32 fp32 x-panel into LDS (async copy when available).
__device__ __forceinline__ void stage_a(const float* xbase, float* dst,
                                        int kt, int wave, int lane) {
  const int rrow = lane >> 3;          // 0..3
  const int kc   = (lane & 7) << 2;    // 0,4,...,28
#pragma unroll
  for (int j = 0; j < 4; ++j) {
    int row = wave * 16 + j * 4 + rrow;                // 8 waves x 16 rows
    const float* g = xbase + (size_t)row * (NMASK * IN_DIM) + kt + kc;
    float* d = dst + row * AFS + kc;
#if HAVE_ASYNC
    __builtin_amdgcn_global_load_async_to_lds_b128((gv4p)g, (lv4p)d, 0, 0);
#else
    *(v4f*)d = *(const v4f*)g;
#endif
  }
}

__device__ __forceinline__ void wait_async0() {
#if HAVE_ASYNC
#if __has_builtin(__builtin_amdgcn_s_wait_asynccnt)
  __builtin_amdgcn_s_wait_asynccnt(0);
#else
  asm volatile("s_wait_asynccnt 0x0" ::: "memory");
#endif
#endif
}

__global__ __launch_bounds__(256, 2)
void masked_gemm_wmma_bf16x3(const float* __restrict__ x,
                             const float* __restrict__ kern,
                             const float* __restrict__ masks,
                             float* __restrict__ out) {
  __shared__ float  Af[2][BM * AFS];          // x panels, fp32, double buffered (36 KB)
  __shared__ __bf16 Bh[BN * BS], Bl[BN * BS]; // (kern.*mask) hi/lo, col-major [col][k] (20 KB)

  const int m  = blockIdx.z;
  const int b0 = blockIdx.y * BM;
  const int n0 = blockIdx.x * BN;

  const int tid  = threadIdx.x;
  const int lane = tid & 31;
  const int wave = tid >> 5;
  const int hf   = lane >> 4;
  const int li   = lane & 15;

  const int wm = (wave & 3) * 32;      // 4 waves along M
  const int wn = (wave >> 2) * 64;     // 2 waves along N

  v8f acc[2][4];
#pragma unroll
  for (int i = 0; i < 2; ++i)
#pragma unroll
    for (int j = 0; j < 4; ++j)
      acc[i][j] = (v8f){0.f, 0.f, 0.f, 0.f, 0.f, 0.f, 0.f, 0.f};

  const float* xbase = x + (size_t)b0 * (NMASK * IN_DIM) + (size_t)m * IN_DIM;
  const float* mbase = masks + (size_t)m * IN_DIM * OUT_DIM;

  // B staging map: k-granule in low lane bits (bank spread), 128B global segments
  const int kg = (tid & 3) | (((tid >> 7) & 1) << 2);  // 0..7
  const int cb = (tid >> 2) & 31;                      // 0..31
  const int k0 = kg * 4;
  const int c0 = cb * 4;

  // Prologue: bring panel 0 in flight.
  stage_a(xbase, &Af[0][0], 0, wave, lane);

  for (int p = 0; p < NPANEL; ++p) {
    const int kt  = p * BK;
    const int buf = p & 1;

    // ---- stage B: (kernel .* mask) -> bf16 hi/lo, 4x4 register transpose ----
    {
      float pr[4][4];
#pragma unroll
      for (int r = 0; r < 4; ++r) {
        size_t g = (size_t)(kt + k0 + r) * OUT_DIM + n0 + c0;
        v4f kv = *(const v4f*)(kern + g);
        v4f mv = *(const v4f*)(mbase + g);
#pragma unroll
        for (int e = 0; e < 4; ++e) pr[r][e] = kv[e] * mv[e];
      }
      if (kt + BK < IN_DIM) {   // hint next panel into cache
        size_t gn = (size_t)(kt + BK + k0) * OUT_DIM + n0 + c0;
        __builtin_prefetch(kern + gn, 0, 1);
        __builtin_prefetch(mbase + gn, 0, 1);
      }
#pragma unroll
      for (int e = 0; e < 4; ++e) {
        v4bf h, l;
#pragma unroll
        for (int r = 0; r < 4; ++r) {
          __bf16 hb = (__bf16)pr[r][e];
          h[r] = hb;
          l[r] = (__bf16)(pr[r][e] - (float)hb);
        }
        *(v4bf*)&Bh[(c0 + e) * BS + k0] = h;
        *(v4bf*)&Bl[(c0 + e) * BS + k0] = l;
      }
    }

    wait_async0();        // panel p's async copy complete (this wave)
    __syncthreads();      // all waves: B stores + async A visible

    // Overlap: launch async copy of panel p+1 while computing panel p.
    if (p + 1 < NPANEL)
      stage_a(xbase, &Af[buf ^ 1][0], kt + BK, wave, lane);

    // ---- compute panel p: split A frags from fp32 LDS, 24 bf16 WMMAs/wave ----
    const float* Ab = &Af[buf][0];
    Frag ah[2], al[2];
#pragma unroll
    for (int i = 0; i < 2; ++i) {
      int row = wm + i * 16 + li;
      const float* pa = Ab + row * AFS + hf * 8;
      split8(pa,      ah[i].u[0], al[i].u[0]);   // K = [8hf, 8hf+8)
      split8(pa + 16, ah[i].u[1], al[i].u[1]);   // K = [16+8hf, 16+8hf+8)
    }
#pragma unroll
    for (int j = 0; j < 4; ++j) {
      int col = wn + j * 16 + li;
      Frag bh, bl;
      const __bf16* pb = &Bh[col * BS + hf * 8];
      bh.u[0] = *(const v4u*)pb;
      bh.u[1] = *(const v4u*)(pb + 16);
      const __bf16* pq = &Bl[col * BS + hf * 8];
      bl.u[0] = *(const v4u*)pq;
      bl.u[1] = *(const v4u*)(pq + 16);
#pragma unroll
      for (int i = 0; i < 2; ++i) {
        acc[i][j] = __builtin_amdgcn_wmma_f32_16x16x32_bf16(
            false, ah[i].bf, false, bh.bf, (short)0, acc[i][j], false, false);
        acc[i][j] = __builtin_amdgcn_wmma_f32_16x16x32_bf16(
            false, al[i].bf, false, bh.bf, (short)0, acc[i][j], false, false);
        acc[i][j] = __builtin_amdgcn_wmma_f32_16x16x32_bf16(
            false, ah[i].bf, false, bl.bf, (short)0, acc[i][j], false, false);
      }
    }

    __syncthreads();      // everyone done reading Bs / Af[buf]
  }

  // ---- epilogue ----
  float* obase = out + (size_t)b0 * (NMASK * OUT_DIM) + (size_t)m * OUT_DIM + n0;
#pragma unroll
  for (int i = 0; i < 2; ++i)
#pragma unroll
    for (int j = 0; j < 4; ++j)
#pragma unroll
      for (int r = 0; r < 8; ++r) {
        int row = wm + i * 16 + r + 8 * hf;
        int col = wn + j * 16 + li;
        obase[(size_t)row * (NMASK * OUT_DIM) + col] = acc[i][j][r];
      }
}

extern "C" void kernel_launch(void* const* d_in, const int* in_sizes, int n_in,
                              void* d_out, int out_size, void* d_ws, size_t ws_size,
                              hipStream_t stream) {
  (void)in_sizes; (void)n_in; (void)out_size; (void)d_ws; (void)ws_size;
  const float* x     = (const float*)d_in[0];
  const float* kern  = (const float*)d_in[1];
  const float* masks = (const float*)d_in[2];
  float* out = (float*)d_out;

  dim3 grid(OUT_DIM / BN, BATCH / BM, NMASK);   // (8, 4, 16)
  dim3 block(256);
  masked_gemm_wmma_bf16x3<<<grid, block, 0, stream>>>(x, kern, masks, out);
}
